// LSTM_63239098466968
// MI455X (gfx1250) — matrix operand order repaired and checked by reading, not compiled, not run on hardware
//
#include <hip/hip_runtime.h>
#include <stdint.h>

typedef __attribute__((ext_vector_type(16))) _Float16 v16h;
typedef __attribute__((ext_vector_type(8)))  _Float16 v8h;
typedef __attribute__((ext_vector_type(2)))  _Float16 h2;
typedef __attribute__((ext_vector_type(8)))  float    v8f;
typedef __attribute__((ext_vector_type(4)))  float    v4f;

#define HID   32     // hidden units
#define GATES 128    // 4*HID, gate order i,f,g,o
#define BT    16     // batch tile == WMMA M
#define LOG2E 1.4426950408889634f

#if defined(__has_builtin)
#if __has_builtin(__builtin_amdgcn_tanhf)
#define HAVE_HW_TANH 1
#endif
#endif

__device__ __forceinline__ float fast_tanh(float x) {
#ifdef HAVE_HW_TANH
    return __builtin_amdgcn_tanhf(x);          // v_tanh_f32 (gfx1250 trans op)
#else
    float e = __builtin_amdgcn_exp2f(2.f * LOG2E * x);
    return 1.f - 2.f * __builtin_amdgcn_rcpf(e + 1.f);
#endif
}
__device__ __forceinline__ float fast_sigm(float x) {
#ifdef HAVE_HW_TANH
    return __builtin_fmaf(0.5f, __builtin_amdgcn_tanhf(0.5f * x), 0.5f);
#else
    return __builtin_amdgcn_rcpf(1.f + __builtin_amdgcn_exp2f(-LOG2E * x));
#endif
}

// Split workgroup barrier. Waits only on DScnt (LDS visibility) -- unlike
// __syncthreads(), it does NOT stall on outstanding global stores (storecnt)
// every timestep. Signal and wait are split so independent work (global
// prefetch, the input-projection WMMA) can overlap the barrier latency.
__device__ __forceinline__ void wg_signal_lds() {
    asm volatile("s_wait_dscnt 0x0\n\ts_barrier_signal -1" ::: "memory");
}
__device__ __forceinline__ void wg_wait() {
    asm volatile("s_barrier_wait -1" ::: "memory");
}

// Convert the three [128][32] f32 weight matrices to f16 (B-fragment wants
// K contiguous per gate column, which is exactly W[g][k] row-major).
__global__ void prep_weights(const float* __restrict__ w0, const float* __restrict__ w1,
                             const float* __restrict__ w2,
                             _Float16* __restrict__ o0, _Float16* __restrict__ o1,
                             _Float16* __restrict__ o2) {
    int i = blockIdx.x * blockDim.x + threadIdx.x;
    if (i < GATES * HID) {
        o0[i] = (_Float16)w0[i];
        o1[i] = (_Float16)w1[i];
        o2[i] = (_Float16)w2[i];
    }
}

// ---------------- Layer 0: IN=1, gates = (x*w_ih + bias) + h @ W_hh^T ----------------
__global__ __launch_bounds__(256) void lstm_layer0(
    const float* __restrict__ x,        // [T,B]
    const float* __restrict__ wih0,     // [128]
    const float* __restrict__ bih0,
    const float* __restrict__ bhh0,
    const _Float16* __restrict__ whh,   // [128][32] f16
    _Float16* __restrict__ h0out,       // [T,B,32] f16
    int T, int B)
{
    __shared__ alignas(16) _Float16 hbuf[BT][HID];
    __shared__ float gates[BT][GATES];
    __shared__ alignas(16) float xbuf[BT];

    const int tid   = threadIdx.x;
    const int lane  = tid & 31;
    const int wave  = tid >> 5;
    const int bbase = blockIdx.x * BT;
    const int ln    = lane & 15;
    const int lh    = lane >> 4;

    const int col  = wave * 16 + ln;
    const int koff = lh * 16;
    const v16h bm = *(const v16h*)(whh + col * HID + koff);

    const float bsum = bih0[col] + bhh0[col];
    const float wcol = wih0[col];

    for (int i = tid; i < BT * HID; i += 256) ((_Float16*)hbuf)[i] = (_Float16)0.f;
    if (tid < BT) xbuf[tid] = x[(size_t)0 * B + bbase + tid];
    float c0 = 0.f, c1 = 0.f;
    const int e0 = tid * 2;
    const int eb = e0 >> 5;
    const int eu = e0 & 31;
    __syncthreads();

    const int arow  = ln;
    const int ahalf = lh * 16;

    for (int t = 0; t < T; ++t) {
        // ---- gates phase ----
        // Fold x*w_ih + bias into the WMMA C operand (pre-WMMA VALU; avoids
        // post-WMMA read hazards before the gate stores).
        v4f xa = *(const v4f*)&xbuf[lh * 8];
        v4f xb = *(const v4f*)&xbuf[lh * 8 + 4];
        v8f cvec;
#pragma unroll
        for (int r = 0; r < 4; ++r) { cvec[r] = __builtin_fmaf(xa[r], wcol, bsum); }
#pragma unroll
        for (int r = 0; r < 4; ++r) { cvec[4 + r] = __builtin_fmaf(xb[r], wcol, bsum); }

        v8h a0 = *(const v8h*)&hbuf[arow][ahalf];
        v8h a1 = *(const v8h*)&hbuf[arow][ahalf + 8];
        v16h a;
#pragma unroll
        for (int i = 0; i < 8; ++i) { a[i] = a0[i]; a[8 + i] = a1[i]; }

        v8f acc = __builtin_amdgcn_wmma_f32_16x16x32_f16(false, a, false, bm,
                                                         (short)0, cvec, false, false);
#pragma unroll
        for (int r = 0; r < 8; ++r) gates[r + lh * 8][col] = acc[r];
        wg_signal_lds();
        wg_wait();

        // ---- elementwise phase (c stays in registers) ----
        {
            float i0  = fast_sigm(gates[eb][eu]);
            float f0  = fast_sigm(gates[eb][eu + 32]);
            float g0  = fast_tanh(gates[eb][eu + 64]);
            float og0 = fast_sigm(gates[eb][eu + 96]);
            c0 = f0 * c0 + i0 * g0;
            float hv0 = og0 * fast_tanh(c0);

            float i1  = fast_sigm(gates[eb][eu + 1]);
            float f1  = fast_sigm(gates[eb][eu + 33]);
            float g1  = fast_tanh(gates[eb][eu + 65]);
            float og1 = fast_sigm(gates[eb][eu + 97]);
            c1 = f1 * c1 + i1 * g1;
            float hv1 = og1 * fast_tanh(c1);

            h2 hp; hp[0] = (_Float16)hv0; hp[1] = (_Float16)hv1;
            *(h2*)&hbuf[eb][eu] = hp;
            *(h2*)&h0out[((size_t)t * B + bbase + eb) * HID + eu] = hp;

            if (tid < BT && t + 1 < T) xbuf[tid] = x[(size_t)(t + 1) * B + bbase + tid];
        }
        wg_signal_lds();
        wg_wait();
    }
}

// ---- Layer 1 + dense head. The input-projection WMMA (global data only) is
// pipelined into the barrier signal->wait window so the per-step critical path
// contains exactly one (recurrent) WMMA. ----
__global__ __launch_bounds__(256) void lstm_layer1_head(
    const _Float16* __restrict__ h0in,  // [T,B,32] f16
    const _Float16* __restrict__ wih,   // [128][32] f16
    const _Float16* __restrict__ whh,   // [128][32] f16
    const float* __restrict__ bih,
    const float* __restrict__ bhh,
    const float* __restrict__ wd,       // [32]
    const float* __restrict__ bd,       // [1]
    float* __restrict__ out,            // [T,B]
    int T, int B)
{
    __shared__ alignas(16) _Float16 hbuf[BT][HID];
    __shared__ float gates[BT][GATES];
    __shared__ float wdbuf[HID];

    const int tid   = threadIdx.x;
    const int lane  = tid & 31;
    const int wave  = tid >> 5;
    const int bbase = blockIdx.x * BT;
    const int ln    = lane & 15;
    const int lh    = lane >> 4;

    const int col  = wave * 16 + ln;
    const int koff = lh * 16;
    const v16h bmx = *(const v16h*)(wih + col * HID + koff);
    const v16h bmh = *(const v16h*)(whh + col * HID + koff);

    const float bsum = bih[col] + bhh[col];
    v8f cbias;
#pragma unroll
    for (int r = 0; r < 8; ++r) cbias[r] = bsum;
    const float bdv = bd[0];

    for (int i = tid; i < BT * HID; i += 256) ((_Float16*)hbuf)[i] = (_Float16)0.f;
    if (tid < HID) wdbuf[tid] = wd[tid];
    float c0 = 0.f, c1 = 0.f;
    const int e0 = tid * 2;
    const int eb = e0 >> 5;
    const int eu = e0 & 31;

    const int arow  = ln;
    const int ahalf = lh * 16;

    // Prime the pipeline: acc1 = h0[0] @ W_ih^T + bias (independent of recurrence)
    const _Float16* hp0 = h0in + ((size_t)0 * B + bbase + arow) * HID;
    v8h px0 = *(const v8h*)(hp0 + ahalf);
    v8h px1 = *(const v8h*)(hp0 + ahalf + 8);
    v16h ax;
#pragma unroll
    for (int i = 0; i < 8; ++i) { ax[i] = px0[i]; ax[8 + i] = px1[i]; }
    v8f acc1 = __builtin_amdgcn_wmma_f32_16x16x32_f16(false, ax, false, bmx,
                                                      (short)0, cbias, false, false);
    __syncthreads();

    for (int t = 0; t < T; ++t) {
        // ---- gates phase: one recurrent WMMA on the critical path ----
        v8h a0 = *(const v8h*)&hbuf[arow][ahalf];
        v8h a1 = *(const v8h*)&hbuf[arow][ahalf + 8];
        v16h ah;
#pragma unroll
        for (int i = 0; i < 8; ++i) { ah[i] = a0[i]; ah[8 + i] = a1[i]; }

        v8f acc = __builtin_amdgcn_wmma_f32_16x16x32_f16(false, ah, false, bmh,
                                                         (short)0, acc1, false, false);
#pragma unroll
        for (int r = 0; r < 8; ++r) gates[r + lh * 8][col] = acc[r];

        // dense head for step t-1: hbuf still holds h1[t-1]; overlapped here,
        // completed before this thread's barrier signal (writers wait on it).
        if (t > 0 && tid < BT) {
            float s = bdv;
            const v8h* hv = (const v8h*)&hbuf[tid][0];
#pragma unroll
            for (int q = 0; q < 4; ++q) {
                v8h hq = hv[q];
#pragma unroll
                for (int j = 0; j < 8; ++j) s += (float)hq[j] * wdbuf[q * 8 + j];
            }
            out[(size_t)(t - 1) * B + bbase + tid] = s;
        }
        wg_signal_lds();

        // ---- overlapped with barrier: next step's input projection ----
        {
            int tn = (t + 1 < T) ? (t + 1) : (T - 1);
            const _Float16* hp = h0in + ((size_t)tn * B + bbase + arow) * HID;
            v8h n0 = *(const v8h*)(hp + ahalf);
            v8h n1 = *(const v8h*)(hp + ahalf + 8);
            v16h axn;
#pragma unroll
            for (int i = 0; i < 8; ++i) { axn[i] = n0[i]; axn[8 + i] = n1[i]; }
            acc1 = __builtin_amdgcn_wmma_f32_16x16x32_f16(false, axn, false, bmx,
                                                          (short)0, cbias, false, false);
        }
        wg_wait();

        // ---- elementwise phase ----
        {
            float i0  = fast_sigm(gates[eb][eu]);
            float f0  = fast_sigm(gates[eb][eu + 32]);
            float g0  = fast_tanh(gates[eb][eu + 64]);
            float og0 = fast_sigm(gates[eb][eu + 96]);
            c0 = f0 * c0 + i0 * g0;
            float hv0 = og0 * fast_tanh(c0);

            float i1  = fast_sigm(gates[eb][eu + 1]);
            float f1  = fast_sigm(gates[eb][eu + 33]);
            float g1  = fast_tanh(gates[eb][eu + 65]);
            float og1 = fast_sigm(gates[eb][eu + 97]);
            c1 = f1 * c1 + i1 * g1;
            float hv1 = og1 * fast_tanh(c1);

            h2 hp2; hp2[0] = (_Float16)hv0; hp2[1] = (_Float16)hv1;
            *(h2*)&hbuf[eb][eu] = hp2;
        }
        wg_signal_lds();
        wg_wait();
    }

    // final dense head: out[T-1] from hbuf = h1[T-1]
    if (tid < BT) {
        float s = bdv;
        const v8h* hv = (const v8h*)&hbuf[tid][0];
#pragma unroll
        for (int q = 0; q < 4; ++q) {
            v8h hq = hv[q];
#pragma unroll
            for (int j = 0; j < 8; ++j) s += (float)hq[j] * wdbuf[q * 8 + j];
        }
        out[(size_t)(T - 1) * B + bbase + tid] = s;
    }
}

extern "C" void kernel_launch(void* const* d_in, const int* in_sizes, int n_in,
                              void* d_out, int out_size, void* d_ws, size_t ws_size,
                              hipStream_t stream) {
    const float* x     = (const float*)d_in[0];
    const float* W_ih0 = (const float*)d_in[1];
    const float* W_hh0 = (const float*)d_in[2];
    const float* b_ih0 = (const float*)d_in[3];
    const float* b_hh0 = (const float*)d_in[4];
    const float* W_ih1 = (const float*)d_in[5];
    const float* W_hh1 = (const float*)d_in[6];
    const float* b_ih1 = (const float*)d_in[7];
    const float* b_hh1 = (const float*)d_in[8];
    const float* W_d   = (const float*)d_in[9];
    const float* b_d   = (const float*)d_in[10];

    const int B = 512;
    const int T = in_sizes[0] / B;

    char* ws = (char*)d_ws;
    _Float16* h0buf = (_Float16*)ws;                         // [T,B,32] f16
    size_t h0bytes  = (size_t)T * B * HID * sizeof(_Float16);
    _Float16* whh0f = (_Float16*)(ws + h0bytes);
    _Float16* wih1f = whh0f + GATES * HID;
    _Float16* whh1f = wih1f + GATES * HID;

    hipLaunchKernelGGL(prep_weights, dim3((GATES * HID + 255) / 256), dim3(256), 0, stream,
                       W_hh0, W_ih1, W_hh1, whh0f, wih1f, whh1f);
    hipLaunchKernelGGL(lstm_layer0, dim3(B / BT), dim3(256), 0, stream,
                       x, W_ih0, b_ih0, b_hh0, whh0f, h0buf, T, B);
    hipLaunchKernelGGL(lstm_layer1_head, dim3(B / BT), dim3(256), 0, stream,
                       h0buf, wih1f, whh1f, b_ih1, b_hh1, W_d, b_d, (float*)d_out, T, B);
}
